// PGC_GNN_68951404970322
// MI455X (gfx1250) — matrix operand order repaired and checked by reading, not compiled, not run on hardware
//
#include <hip/hip_runtime.h>
#include <hip/hip_bf16.h>

#define FN 50000      // nodes
#define NE 600000     // edges
#define NG 128        // graphs
#define C0 128        // IN_CH / PROJ / OUT_CH
#define CM 384        // OUT_CH*K
#define C3 1152       // 3*CM
#define H1 768        // 2*CM
#define PB 128        // nodes per pooling block

typedef __attribute__((ext_vector_type(16))) __bf16 bf16x16;
typedef __attribute__((ext_vector_type(8)))  __bf16 bf16x8;
typedef __attribute__((ext_vector_type(8)))  float  floatx8;

// ---------------- atomics ----------------
__device__ __forceinline__ void atomAddF(float* p, float v) {
  __hip_atomic_fetch_add(p, v, __ATOMIC_RELAXED, __HIP_MEMORY_SCOPE_AGENT);
}
__device__ __forceinline__ void atomMaxF(float* p, float v) {
  // single-hw-atomic float max trick (works with -3e38 init)
  if (v >= 0.0f) atomicMax((int*)p, __float_as_int(v));
  else           atomicMin((unsigned int*)p, (unsigned int)__float_as_int(v));
}

// ---------------- utility kernels ----------------
__global__ void k_fill(float* __restrict__ p, int n, float v) {
  int i = blockIdx.x * blockDim.x + threadIdx.x;
  if (i < n) p[i] = v;
}

__global__ void k_prep_conv_w(const float* __restrict__ wrel, const float* __restrict__ wroot,
                              __bf16* __restrict__ brel, __bf16* __restrict__ broot) {
  int i = blockIdx.x * blockDim.x + threadIdx.x;
  if (i < C0 * C0) { brel[i] = (__bf16)wrel[i]; broot[i] = (__bf16)wroot[i]; }
}

__global__ void k_prep_linw(const float* __restrict__ w, __bf16* __restrict__ out) {
  int i = blockIdx.x * blockDim.x + threadIdx.x;
  if (i < CM * CM) {
    int r = i / CM, c = i % CM;
    float v = (c < C0 * ((r >> 7) + 1)) ? w[i] : 0.0f;  // xhi block-causal mask
    out[i] = (__bf16)v;
  }
}

// ---------------- edge scatter-add SpMM: dst[sidx[e]] += src[gidx[e]] (128 ch) ----------------
__global__ void k_edge_spmm(const int* __restrict__ gidx, const int* __restrict__ sidx,
                            const float* __restrict__ src, float* __restrict__ dst) {
  long t = (long)blockIdx.x * blockDim.x + threadIdx.x;
  int e = (int)(t >> 5);
  if (e >= NE) return;
  int lane = (int)t & 31;
  int s = gidx[e], d = sidx[e];
  const float4 v = *(const float4*)(src + (size_t)s * C0 + lane * 4);
  float* o = dst + (size_t)d * C0 + lane * 4;
  atomAddF(o + 0, v.x); atomAddF(o + 1, v.y);
  atomAddF(o + 2, v.z); atomAddF(o + 3, v.w);
}

// ---------------- WMMA fragment loaders ----------------
// A 16x32 bf16 tile: lane holds row (lane&15); elem e -> K = hi*16 + e
__device__ __forceinline__ bf16x16 load_a_f32(const float* __restrict__ rowp, int hi) {
  const float* q0 = rowp + hi * 16;
  const float* q1 = rowp + hi * 16 + 8;
  float4 a0 = *(const float4*)q0, a1 = *(const float4*)(q0 + 4);
  float4 a2 = *(const float4*)q1, a3 = *(const float4*)(q1 + 4);
  bf16x16 a;
  a[0]  = (__bf16)a0.x; a[1]  = (__bf16)a0.y; a[2]  = (__bf16)a0.z; a[3]  = (__bf16)a0.w;
  a[4]  = (__bf16)a1.x; a[5]  = (__bf16)a1.y; a[6]  = (__bf16)a1.z; a[7]  = (__bf16)a1.w;
  a[8]  = (__bf16)a2.x; a[9]  = (__bf16)a2.y; a[10] = (__bf16)a2.z; a[11] = (__bf16)a2.w;
  a[12] = (__bf16)a3.x; a[13] = (__bf16)a3.y; a[14] = (__bf16)a3.z; a[15] = (__bf16)a3.w;
  return a;
}
__device__ __forceinline__ bf16x16 load_a_bf(const __bf16* __restrict__ rowp, int hi) {
  bf16x8 lo = *(const bf16x8*)(rowp + hi * 16);
  bf16x8 hv = *(const bf16x8*)(rowp + hi * 16 + 8);
  return __builtin_shufflevector(lo, hv, 0,1,2,3,4,5,6,7,8,9,10,11,12,13,14,15);
}
// B 32x16 bf16 tile: lane holds col (lane&15); elem e -> K = hi*16 + e (one 32B load)
__device__ __forceinline__ bf16x16 load_b_bf(const __bf16* __restrict__ p) {
  return *(const bf16x16*)p;
}

// ---------------- GraphConv GEMM: X = neigh @ Wrel^T + x @ Wroot^T ----------------
// 10 waves/block (320 thr), grid 625 -> 6250 waves = 3125 M-tiles x 2 N-groups.
// Wave: 1 M-tile x 4 N-tiles. k-loop flattened into 8 stages alternating
// (neigh x Wrel, x x Wroot) with one-stage-ahead double-buffered fragments.
__global__ void k_conv_gemm(const float* __restrict__ neigh, const float* __restrict__ x,
                            const __bf16* __restrict__ wrel, const __bf16* __restrict__ wroot,
                            float* __restrict__ out) {
  int wid = threadIdx.x >> 5;
  int mTile = blockIdx.x * 5 + (wid >> 1);
  int nBase = (wid & 1) * 4;
  int lane = threadIdx.x & 31;
  int hi = lane >> 4, l15 = lane & 15;
  int arow = mTile * 16 + l15;

  const float* srcs[2] = {neigh, x};
  const __bf16* wsrc[2] = {wrel, wroot};

  const floatx8 zero = {0.f, 0.f, 0.f, 0.f, 0.f, 0.f, 0.f, 0.f};
  floatx8 acc[4] = {zero, zero, zero, zero};

  // prologue: stage 0 fragments
  bf16x16 a = load_a_f32(neigh + (size_t)arow * C0, hi);
  bf16x16 b[4];
#pragma unroll
  for (int n = 0; n < 4; n++)
    b[n] = load_b_bf(wrel + (size_t)((nBase + n) * 16 + l15) * C0 + hi * 16);

#pragma unroll
  for (int s = 0; s < 8; s++) {   // stage s: k0=(s>>1)*32, matrix=(s&1)
    bf16x16 an, bn[4];
    if (s < 7) {
      int s1 = s + 1;
      int k0n = (s1 >> 1) * 32;
      const float* sp = srcs[s1 & 1];
      const __bf16* wp = wsrc[s1 & 1];
      an = load_a_f32(sp + (size_t)arow * C0 + k0n, hi);
#pragma unroll
      for (int n = 0; n < 4; n++)
        bn[n] = load_b_bf(wp + (size_t)((nBase + n) * 16 + l15) * C0 + k0n + hi * 16);
    }
#pragma unroll
    for (int n = 0; n < 4; n++)
      acc[n] = __builtin_amdgcn_wmma_f32_16x16x32_bf16(false, a, false, b[n], (short)0, acc[n], false, false);
    if (s < 7) {
      a = an;
#pragma unroll
      for (int n = 0; n < 4; n++) b[n] = bn[n];
    }
  }

#pragma unroll
  for (int n = 0; n < 4; n++)
#pragma unroll
    for (int r = 0; r < 8; r++)
      out[(size_t)(mTile * 16 + r + 8 * hi) * C0 + (nBase + n) * 16 + l15] = acc[n][r];
}

// ---------------- big masked GEMM: Y = Hc(bf16)[N,384] @ linw(bf16)[384,384]^T ----------------
// 6 waves/block (192 thr), grid 3125; wave w: 1 M-tile x 4 N-tiles; double-buffered k-loop.
__global__ void k_big_gemm(const __bf16* __restrict__ A, const __bf16* __restrict__ W,
                           float* __restrict__ Y) {
  int mTile = blockIdx.x;
  int nBase = (threadIdx.x >> 5) * 4;
  int lane = threadIdx.x & 31;
  int hi = lane >> 4, l15 = lane & 15;
  const __bf16* aRow = A + (size_t)(mTile * 16 + l15) * CM;
  const __bf16* wr[4];
#pragma unroll
  for (int n = 0; n < 4; n++)
    wr[n] = W + (size_t)((nBase + n) * 16 + l15) * CM + hi * 16;

  const floatx8 zero = {0.f, 0.f, 0.f, 0.f, 0.f, 0.f, 0.f, 0.f};
  floatx8 acc[4] = {zero, zero, zero, zero};

  bf16x16 a = load_a_bf(aRow, hi);
  bf16x16 b[4];
#pragma unroll
  for (int n = 0; n < 4; n++) b[n] = load_b_bf(wr[n]);

#pragma unroll
  for (int k0 = 0; k0 < CM; k0 += 32) {
    bf16x16 an, bn[4];
    if (k0 + 32 < CM) {
      an = load_a_bf(aRow + k0 + 32, hi);
#pragma unroll
      for (int n = 0; n < 4; n++) bn[n] = load_b_bf(wr[n] + k0 + 32);
    }
#pragma unroll
    for (int n = 0; n < 4; n++)
      acc[n] = __builtin_amdgcn_wmma_f32_16x16x32_bf16(false, a, false, b[n], (short)0, acc[n], false, false);
    if (k0 + 32 < CM) {
      a = an;
#pragma unroll
      for (int n = 0; n < 4; n++) b[n] = bn[n];
    }
  }

#pragma unroll
  for (int n = 0; n < 4; n++)
#pragma unroll
    for (int r = 0; r < 8; r++)
      Y[(size_t)(mTile * 16 + r + 8 * hi) * CM + (nBase + n) * 16 + l15] = acc[n][r];
}

// ---------------- per-channel batch stats (coalesced: blockDim == C) ----------------
__global__ void k_stats_acc(const float* __restrict__ Y, int C,
                            float* __restrict__ s1, float* __restrict__ s2g) {
  int c = threadIdx.x;  // blockDim == C -> consecutive lanes, consecutive addresses
  float s = 0.f, s2 = 0.f;
  for (int r = blockIdx.x; r < FN; r += gridDim.x) {
    float v = Y[(size_t)r * C + c];
    s += v; s2 += v * v;
  }
  atomAddF(&s1[c], s);
  atomAddF(&s2g[c], s2);
}
__global__ void k_stats_fin(const float* __restrict__ s1, const float* __restrict__ s2g, int C,
                            float* __restrict__ mean, float* __restrict__ inv) {
  int c = blockIdx.x * blockDim.x + threadIdx.x;
  if (c >= C) return;
  float m = s1[c] / (float)FN;
  float var = s2g[c] / (float)FN - m * m;
  mean[c] = m;
  inv[c] = rsqrtf(fmaxf(var, 0.f) + 1e-5f);
}

// BN apply to X (in place) + write bf16 into Hc slice 0
__global__ void k_bn0(float* __restrict__ X, const float* __restrict__ mean,
                      const float* __restrict__ inv, const float* __restrict__ g,
                      const float* __restrict__ b, __bf16* __restrict__ Hc) {
  int i = blockIdx.x * 256 + threadIdx.x;
  if (i >= FN * C0) return;
  int c = i & 127, r = i >> 7;
  float v = (X[i] - mean[c]) * inv[c] * g[c] + b[c];
  X[i] = v;
  Hc[(size_t)r * CM + c] = (__bf16)v;
}

__global__ void k_toslice(const float* __restrict__ src, __bf16* __restrict__ Hc, int off) {
  int i = blockIdx.x * 256 + threadIdx.x;
  if (i >= FN * C0) return;
  int c = i & 127, r = i >> 7;
  Hc[(size_t)r * CM + off + c] = (__bf16)src[i];
}

__global__ void k_bnh(float* __restrict__ Y, const float* __restrict__ mean,
                      const float* __restrict__ inv, const float* __restrict__ g,
                      const float* __restrict__ b) {
  int i = blockIdx.x * 256 + threadIdx.x; // FN*CM = 19.2M < 2^31
  if (i >= FN * CM) return;
  int c = i % CM;
  Y[i] = (Y[i] - mean[c]) * inv[c] * g[c] + b[c];
}

// ---------------- pooling ----------------
__global__ void k_count(const int* __restrict__ batch, float* __restrict__ cnt) {
  int i = blockIdx.x * blockDim.x + threadIdx.x;
  if (i < FN) atomAddF(&cnt[batch[i]], 1.0f);
}

__device__ __forceinline__ void pool_flush(int g, int c4, float4 s, float4 mx,
                                           float* __restrict__ Hadd, float* __restrict__ Hmax) {
  float* pa = Hadd + (size_t)g * CM + c4;
  float* pm = Hmax + (size_t)g * CM + c4;
  atomAddF(pa + 0, s.x); atomAddF(pa + 1, s.y); atomAddF(pa + 2, s.z); atomAddF(pa + 3, s.w);
  atomMaxF(pm + 0, mx.x); atomMaxF(pm + 1, mx.y); atomMaxF(pm + 2, mx.z); atomMaxF(pm + 3, mx.w);
}

// batch is sorted: walk PB consecutive nodes per block, flush atomics only on graph boundary.
__global__ void k_pool(const float* __restrict__ Hm, const int* __restrict__ batch,
                       float* __restrict__ Hadd, float* __restrict__ Hmax) {
  int n0 = blockIdx.x * PB;
  int c4 = threadIdx.x * 4;  // 96 threads -> 384 channels
  int nEnd = (n0 + PB < FN) ? (n0 + PB) : FN;
  const float4 mInit = {-3.0e38f, -3.0e38f, -3.0e38f, -3.0e38f};
  float4 s = {0.f, 0.f, 0.f, 0.f};
  float4 mx = mInit;
  int gcur = batch[n0];
  for (int n = n0; n < nEnd; n++) {
    int g = batch[n];  // uniform across block -> no divergence
    if (g != gcur) {
      pool_flush(gcur, c4, s, mx, Hadd, Hmax);
      s = (float4){0.f, 0.f, 0.f, 0.f};
      mx = mInit;
      gcur = g;
    }
    const float4 v = *(const float4*)(Hm + (size_t)n * CM + c4);
    s.x += v.x; s.y += v.y; s.z += v.z; s.w += v.w;
    mx.x = fmaxf(mx.x, v.x); mx.y = fmaxf(mx.y, v.y);
    mx.z = fmaxf(mx.z, v.z); mx.w = fmaxf(mx.w, v.w);
  }
  pool_flush(gcur, c4, s, mx, Hadd, Hmax);
}

// ---------------- readout ----------------
__device__ __forceinline__ float hcat_fetch(const float* Hadd, const float* Hmax,
                                            const float* cnt, int g, int c) {
  if (c < CM)      return Hadd[(size_t)g * CM + c] / fmaxf(cnt[g], 1.0f);
  if (c < 2 * CM)  return Hadd[(size_t)g * CM + (c - CM)];
  float v = Hmax[(size_t)g * CM + (c - 2 * CM)];
  return (v < -1e37f) ? 0.0f : v;   // isfinite guard for empty segments
}

__global__ void k_bno(const float* __restrict__ Hadd, const float* __restrict__ Hmax,
                      const float* __restrict__ cnt, const float* __restrict__ g,
                      const float* __restrict__ b, float* __restrict__ Hbn) {
  int c = blockIdx.x * blockDim.x + threadIdx.x;
  if (c >= C3) return;
  float s = 0.f, s2 = 0.f;
  for (int gg = 0; gg < NG; gg++) {
    float v = hcat_fetch(Hadd, Hmax, cnt, gg, c);
    s += v; s2 += v * v;
  }
  float m = s / (float)NG;
  float var = s2 / (float)NG - m * m;
  float iv = rsqrtf(fmaxf(var, 0.f) + 1e-5f);
  for (int gg = 0; gg < NG; gg++) {
    float v = hcat_fetch(Hadd, Hmax, cnt, gg, c);
    Hbn[(size_t)gg * C3 + c] = (v - m) * iv * g[c] + b[c];
  }
}

__global__ void k_mlp1(const float* __restrict__ Hbn, const float* __restrict__ w1,
                       const float* __restrict__ b1, float* __restrict__ h1) {
  __shared__ float sh[C3];
  int g = blockIdx.x;
  for (int k = threadIdx.x; k < C3; k += 256) sh[k] = Hbn[(size_t)g * C3 + k];
  __syncthreads();
  for (int o = threadIdx.x; o < H1; o += 256) {
    const float* wr = w1 + (size_t)o * C3;
    float s = 0.f;
    for (int k = 0; k < C3; k++) s += sh[k] * wr[k];
    h1[(size_t)g * H1 + o] = fmaxf(s + b1[o], 0.0f);
  }
}

__global__ void k_mlp2(const float* __restrict__ h1, const float* __restrict__ w2,
                       const float* __restrict__ b2, float* __restrict__ h2) {
  __shared__ float sh[H1];
  int g = blockIdx.x;
  for (int k = threadIdx.x; k < H1; k += 256) sh[k] = h1[(size_t)g * H1 + k];
  __syncthreads();
  for (int o = threadIdx.x; o < CM; o += 256) {
    const float* wr = w2 + (size_t)o * H1;
    float s = 0.f;
    for (int k = 0; k < H1; k++) s += sh[k] * wr[k];
    h2[(size_t)g * CM + o] = fmaxf(s + b2[o], 0.0f);
  }
}

__global__ void k_out(const float* __restrict__ h2, const float* __restrict__ w3,
                      const float* __restrict__ b3, float* __restrict__ out) {
  int g = threadIdx.x;
  if (g >= NG) return;
  const float* hr = h2 + (size_t)g * CM;
  float l0 = b3[0], l1 = b3[1];
  for (int k = 0; k < CM; k++) {
    float h = hr[k];
    l0 += h * w3[k];
    l1 += h * w3[CM + k];
  }
  float m = fmaxf(l0, l1);
  float lse = m + logf(expf(l0 - m) + expf(l1 - m));
  out[g * 2 + 0] = l0 - lse;
  out[g * 2 + 1] = l1 - lse;
}

// ---------------- host launcher ----------------
extern "C" void kernel_launch(void* const* d_in, const int* in_sizes, int n_in,
                              void* d_out, int out_size, void* d_ws, size_t ws_size,
                              hipStream_t stream) {
  const float* x      = (const float*)d_in[0];
  const int*   ei     = (const int*)d_in[1];
  const int*   row    = ei;
  const int*   col    = ei + NE;
  const int*   batch  = (const int*)d_in[2];
  const float* w_rel  = (const float*)d_in[4];
  const float* w_root = (const float*)d_in[6];
  const float* n0g    = (const float*)d_in[7];
  const float* n0b    = (const float*)d_in[8];
  const float* lin_w  = (const float*)d_in[9];
  const float* bnhg   = (const float*)d_in[11];
  const float* bnhb   = (const float*)d_in[12];
  const float* bnog   = (const float*)d_in[13];
  const float* bnob   = (const float*)d_in[14];
  const float* w1     = (const float*)d_in[15];
  const float* b1     = (const float*)d_in[16];
  const float* w2     = (const float*)d_in[17];
  const float* b2     = (const float*)d_in[18];
  const float* w3     = (const float*)d_in[19];
  const float* b3     = (const float*)d_in[20];
  float* out = (float*)d_out;

  char* ws = (char*)d_ws;
  size_t cur = 0;
  auto alloc = [&](size_t bytes) -> char* {
    char* p = ws + cur;
    cur = (cur + bytes + 255) & ~(size_t)255;
    return p;
  };
  float*  bufA   = (float*)alloc(sizeof(float) * FN * C0);   // neigh / hop2
  float*  bufB   = (float*)alloc(sizeof(float) * FN * C0);   // X / X_bn
  float*  bufC   = (float*)alloc(sizeof(float) * FN * C0);   // hop1
  __bf16* Hc     = (__bf16*)alloc(2ULL * FN * CM);           // concat hops (bf16)
  float*  Y      = (float*)alloc(sizeof(float) * (size_t)FN * CM); // Hm (in place)
  __bf16* wrelb  = (__bf16*)alloc(2ULL * C0 * C0);
  __bf16* wrootb = (__bf16*)alloc(2ULL * C0 * C0);
  __bf16* linwb  = (__bf16*)alloc(2ULL * CM * CM);
  float*  statS  = (float*)alloc(sizeof(float) * CM);
  float*  statS2 = (float*)alloc(sizeof(float) * CM);
  float*  mean0  = (float*)alloc(sizeof(float) * C0);
  float*  inv0   = (float*)alloc(sizeof(float) * C0);
  float*  meanh  = (float*)alloc(sizeof(float) * CM);
  float*  invh   = (float*)alloc(sizeof(float) * CM);
  float*  Hadd   = (float*)alloc(sizeof(float) * NG * CM);
  float*  Hmax   = (float*)alloc(sizeof(float) * NG * CM);
  float*  cnt    = (float*)alloc(sizeof(float) * NG);
  float*  Hbn    = (float*)alloc(sizeof(float) * NG * C3);
  float*  h1     = (float*)alloc(sizeof(float) * NG * H1);
  float*  h2     = (float*)alloc(sizeof(float) * NG * CM);

  const int TB = 256;
  const int nNodeF = FN * C0;            // 6,400,000 (divisible by 256)
  const int gNodeF = nNodeF / TB;        // 25000
  const int gEdge  = (NE * 32) / TB;     // 75000
  const int gBnh   = (FN * CM) / TB;     // 75000

  // GraphConv: neigh[col] += x[row]
  k_fill<<<gNodeF, TB, 0, stream>>>(bufA, nNodeF, 0.0f);
  k_prep_conv_w<<<(C0 * C0 + TB - 1) / TB, TB, 0, stream>>>(w_rel, w_root, wrelb, wrootb);
  k_prep_linw<<<(CM * CM + TB - 1) / TB, TB, 0, stream>>>(lin_w, linwb);
  k_edge_spmm<<<gEdge, TB, 0, stream>>>(row, col, x, bufA);
  k_conv_gemm<<<625, 320, 0, stream>>>(bufA, x, wrelb, wrootb, bufB);
  k_fill<<<1, TB, 0, stream>>>(statS, C0, 0.0f);
  k_fill<<<1, TB, 0, stream>>>(statS2, C0, 0.0f);
  k_stats_acc<<<125, C0, 0, stream>>>(bufB, C0, statS, statS2);
  k_stats_fin<<<1, C0, 0, stream>>>(statS, statS2, C0, mean0, inv0);
  k_bn0<<<gNodeF, TB, 0, stream>>>(bufB, mean0, inv0, n0g, n0b, Hc);

  // K-hop diffusion: next[row] += cur[col]
  k_fill<<<gNodeF, TB, 0, stream>>>(bufC, nNodeF, 0.0f);
  k_edge_spmm<<<gEdge, TB, 0, stream>>>(col, row, bufB, bufC);
  k_toslice<<<gNodeF, TB, 0, stream>>>(bufC, Hc, C0);
  k_fill<<<gNodeF, TB, 0, stream>>>(bufA, nNodeF, 0.0f);
  k_edge_spmm<<<gEdge, TB, 0, stream>>>(col, row, bufC, bufA);
  k_toslice<<<gNodeF, TB, 0, stream>>>(bufA, Hc, 2 * C0);

  // masked linear + BN
  k_big_gemm<<<3125, 192, 0, stream>>>(Hc, linwb, Y);
  k_fill<<<2, TB, 0, stream>>>(statS, CM, 0.0f);
  k_fill<<<2, TB, 0, stream>>>(statS2, CM, 0.0f);
  k_stats_acc<<<125, CM, 0, stream>>>(Y, CM, statS, statS2);
  k_stats_fin<<<2, TB, 0, stream>>>(statS, statS2, CM, meanh, invh);
  k_bnh<<<gBnh, TB, 0, stream>>>(Y, meanh, invh, bnhg, bnhb);

  // pooling (batch sorted -> segmented register accumulation)
  k_fill<<<(NG * CM + TB - 1) / TB, TB, 0, stream>>>(Hadd, NG * CM, 0.0f);
  k_fill<<<(NG * CM + TB - 1) / TB, TB, 0, stream>>>(Hmax, NG * CM, -3.0e38f);
  k_fill<<<1, TB, 0, stream>>>(cnt, NG, 0.0f);
  k_count<<<(FN + TB - 1) / TB, TB, 0, stream>>>(batch, cnt);
  k_pool<<<(FN + PB - 1) / PB, 96, 0, stream>>>(Y, batch, Hadd, Hmax);

  // readout MLP
  k_bno<<<(C3 + TB - 1) / TB, TB, 0, stream>>>(Hadd, Hmax, cnt, bnog, bnob, Hbn);
  k_mlp1<<<NG, TB, 0, stream>>>(Hbn, w1, b1, h1);
  k_mlp2<<<NG, TB, 0, stream>>>(h1, w2, b2, h2);
  k_out<<<1, NG, 0, stream>>>(h2, w3, b3, out);
}
